// TACEDescriptor_35777077575993
// MI455X (gfx1250) — matrix operand order repaired and checked by reading, not compiled, not run on hardware
//
#include <hip/hip_runtime.h>
#include <math.h>

typedef _Float16 half_t;
typedef __attribute__((ext_vector_type(16))) _Float16 v16h;
typedef __attribute__((ext_vector_type(8)))  _Float16 v8h;
typedef __attribute__((ext_vector_type(8)))  float    v8f;

#define C_CH   64
#define NBASIS 8
#define RCUT   6.0f
#define AVG_INV 0.0625f
#define ZMAX_  10
#define RWN    192        // radial output cols actually consumed (ranks 0..2)
#define ADIM   13         // 1 + 3 + 9  (rank-3 A is dead code in the reference)
#define PI_F   3.14159265358979323846f

// ---------- small prep kernels ----------

__global__ void k_transpose_f16(const float* __restrict__ src, half_t* __restrict__ dst,
                                int K, int Nn) {
  int i = blockIdx.x * blockDim.x + threadIdx.x;
  if (i >= K * Nn) return;
  int k = i / Nn, n = i - k * Nn;
  dst[(size_t)n * K + k] = (half_t)src[i];
}

__global__ void k_f32_to_f16(const float* __restrict__ s, half_t* __restrict__ d, int n) {
  int i = blockIdx.x * blockDim.x + threadIdx.x;
  if (i < n) d[i] = (half_t)s[i];
}

__global__ void k_zero(float* __restrict__ p, long long n) {
  long long i = (long long)blockIdx.x * blockDim.x + threadIdx.x;
  if (i < n) p[i] = 0.0f;
}

__global__ void k_init_h0(const int* __restrict__ species, const float* __restrict__ W_emb,
                          float* __restrict__ h0, int N) {
  int i = blockIdx.x * blockDim.x + threadIdx.x;
  if (i >= N * C_CH) return;
  int n = i >> 6, c = i & 63;
  h0[i] = W_emb[species[n] * C_CH + c];
}

// ---------- edge geometry: unit vectors + Bessel*cutoff basis ----------

__global__ void k_edge_geom(const float* __restrict__ ev, float* __restrict__ u,
                            float* __restrict__ ef, int E) {
  int e = blockIdx.x * blockDim.x + threadIdx.x;
  if (e >= E) return;
  float x = ev[3*e], y = ev[3*e+1], z = ev[3*e+2];
  float r = sqrtf(x*x + y*y + z*z);
  float inv = 1.0f / r;
  u[3*e] = x*inv; u[3*e+1] = y*inv; u[3*e+2] = z*inv;
  float xr = r / RCUT;
  float fc = 0.0f;
  if (xr < 1.0f) {
    float x2 = xr*xr; float x4 = x2*x2; float x5 = x4*xr;
    fc = 1.0f - 21.0f*x5 + 35.0f*x5*xr - 15.0f*x5*x2;
  }
  float pref = sqrtf(2.0f / RCUT) * inv * fc;
  float w = PI_F * r / RCUT;
  #pragma unroll
  for (int n = 1; n <= NBASIS; n++)
    ef[(size_t)e*NBASIS + n - 1] = pref * sinf((float)n * w);
}

// ---------- radial MLP hidden layer (K=8 -> VALU) ----------

__global__ void k_radial_hidden(const float* __restrict__ ef, const float* __restrict__ w1,
                                half_t* __restrict__ hidden, int E) {
  int i = blockIdx.x * blockDim.x + threadIdx.x;
  if (i >= E * C_CH) return;
  int e = i >> 6, j = i & 63;
  const float* f = ef + (size_t)e * NBASIS;
  float acc = 0.0f;
  #pragma unroll
  for (int k = 0; k < NBASIS; k++) acc += f[k] * w1[k * C_CH + j];
  float s = acc / (1.0f + expf(-acc));            // silu
  hidden[i] = (half_t)s;
}

// ---------- generic WMMA GEMM: C[M,Nn](f32) = A[M,K](f16) * Bt[Nn,K](f16)^T ----------
// One 16x16 output tile per wave; K multiple of 32; M,Nn multiples of 16 (padded).

__global__ void k_gemm_wmma(const half_t* __restrict__ A, const half_t* __restrict__ Bt,
                            float* __restrict__ Cm, int M, int Nn, int K) {
  int wave = blockIdx.x * (blockDim.x >> 5) + (threadIdx.x >> 5);
  int lane = threadIdx.x & 31;
  int ntn  = Nn >> 4;
  int tm   = wave / ntn;
  int tn   = wave - tm * ntn;
  if (tm * 16 >= M) return;                        // wave-uniform guard (EXEC stays full)
  int rc = lane & 15;                              // A-row / B-col within tile
  int hi = lane >> 4;                              // K half selector
  const half_t* arow = A  + (size_t)(tm*16 + rc) * K;
  const half_t* brow = Bt + (size_t)(tn*16 + rc) * K;
  v8f acc = {};
  for (int k0 = 0; k0 < K; k0 += 32) {
    // A-layout: lanes 0-15 hold K k0..k0+15, lanes 16-31 hold k0+16..k0+31 (matches B)
    v16h a = *(const v16h*)(arow + k0 + hi*16);
    // B-layout (same contiguous k order as A): lanes 0-15 hold K k0..k0+15, lanes 16-31 hold k0+16..k0+31
    v16h b = *(const v16h*)(brow + k0 + hi*16);
    acc = __builtin_amdgcn_wmma_f32_16x16x32_f16(false, a, false, b,
                                                 (short)0, acc, false, false);
  }
  // D layout: VGPR j holds M = hi*8 + j, N = lane&15
  float* crow = Cm + (size_t)(tm*16 + hi*8) * Nn + tn*16 + rc;
  #pragma unroll
  for (int j = 0; j < 8; j++) crow[(size_t)j * Nn] = acc[j];
}

// ---------- message construction + scatter-add into A[N,C,13] ----------

__global__ void k_message(const float* __restrict__ rw, const float* __restrict__ sgate,
                          const float* __restrict__ h1, const float* __restrict__ h2,
                          const float* __restrict__ u, const int* __restrict__ ei,
                          float* __restrict__ Aacc, int E, int hasHigher) {
  int i = blockIdx.x * blockDim.x + threadIdx.x;
  int e = i >> 6, c = i & 63;
  if (e >= E) return;
  int src = ei[e], dst = ei[E + e];
  float uv[3] = { u[3*e], u[3*e+1], u[3*e+2] };
  float y2[9];
  #pragma unroll
  for (int a = 0; a < 3; a++)
    #pragma unroll
    for (int b = 0; b < 3; b++)
      y2[a*3+b] = uv[a] * uv[b];
  // density-trick scalar gate: s_e = s[src] + sum_l <h_l[src], Y_l>
  float se = sgate[(size_t)src*C_CH + c];
  if (hasHigher) {
    const float* p1 = h1 + ((size_t)src*C_CH + c)*3;
    se += p1[0]*uv[0] + p1[1]*uv[1] + p1[2]*uv[2];
    const float* p2 = h2 + ((size_t)src*C_CH + c)*9;
    #pragma unroll
    for (int k = 0; k < 9; k++) se += p2[k]*y2[k];
  }
  const float* rwe = rw + (size_t)e*RWN;
  float base = se * AVG_INV;
  float* ap = Aacc + ((size_t)dst*C_CH + c)*ADIM;
  float w0 = rwe[c] * base;
  atomicAdd(ap, w0);
  float w1 = rwe[64 + c] * base;
  #pragma unroll
  for (int k = 0; k < 3; k++) atomicAdd(ap + 1 + k, w1 * uv[k]);
  float w2 = rwe[128 + c] * base;
  #pragma unroll
  for (int k = 0; k < 9; k++) atomicAdd(ap + 4 + k, w2 * y2[k]);
  // rank-3 A is never consumed by the self-contraction -> skipped entirely
}

// ---------- node update: self-contraction (product basis) ----------
// 4 nodes per block; A tile staged in LDS (4*64*13*4 = 13.3 KB).

__global__ void k_node_update(const float* __restrict__ Aacc, const float* __restrict__ h0,
                              const int* __restrict__ species, const float* __restrict__ Wsc,
                              const float* __restrict__ Wmix, float* __restrict__ h0n,
                              float* __restrict__ h1n, float* __restrict__ h2n,
                              float* __restrict__ outDesc, int N, int writeHigher, int descOff) {
  __shared__ float As[4][C_CH][ADIM];
  int local = threadIdx.x >> 6;
  int d = threadIdx.x & 63;
  int n = blockIdx.x * 4 + local;
  if (n < N) {
    const float* ap = Aacc + ((size_t)n*C_CH + d)*ADIM;
    #pragma unroll
    for (int k = 0; k < ADIM; k++) As[local][d][k] = ap[k];
  }
  __syncthreads();
  if (n >= N) return;
  const float* W0 = Wmix;
  const float* W1 = Wmix + C_CH*C_CH;
  const float* W2 = Wmix + 2*C_CH*C_CH;
  float a0 = As[local][d][0];
  float a1[3], a2[9];
  #pragma unroll
  for (int k = 0; k < 3; k++) a1[k] = As[local][d][1+k];
  #pragma unroll
  for (int k = 0; k < 9; k++) a2[k] = As[local][d][4+k];

  float o0 = 0.0f;
  for (int cc = 0; cc < C_CH; cc++) o0 += As[local][cc][0] * W0[cc*C_CH + d];
  #pragma unroll
  for (int k = 0; k < 3; k++) o0 += a1[k]*a1[k];
  #pragma unroll
  for (int k = 0; k < 9; k++) o0 += a2[k]*a2[k];
  o0 += Wsc[species[n]*C_CH + d] * h0[(size_t)n*C_CH + d];   // species self-connection
  h0n[(size_t)n*C_CH + d] = o0;
  outDesc[(size_t)n*(2*C_CH) + descOff + d] = o0;

  if (writeHigher) {
    float r1[3];
    #pragma unroll
    for (int k = 0; k < 3; k++) {
      float acc = 0.0f;
      for (int cc = 0; cc < C_CH; cc++) acc += As[local][cc][1+k] * W1[cc*C_CH + d];
      r1[k] = acc + a0*a1[k];
    }
    #pragma unroll
    for (int ii = 0; ii < 3; ii++) {
      float acc = 0.0f;
      #pragma unroll
      for (int j = 0; j < 3; j++) acc += a2[ii*3+j]*a1[j];
      r1[ii] += acc;
    }
    float* p1 = h1n + ((size_t)n*C_CH + d)*3;
    #pragma unroll
    for (int k = 0; k < 3; k++) p1[k] = r1[k];

    float* p2 = h2n + ((size_t)n*C_CH + d)*9;
    #pragma unroll
    for (int kk = 0; kk < 9; kk++) {
      float acc = 0.0f;
      for (int cc = 0; cc < C_CH; cc++) acc += As[local][cc][4+kk] * W2[cc*C_CH + d];
      acc += a1[kk/3]*a1[kk%3] + a0*a2[kk];
      p2[kk] = acc;
    }
  }
}

// ---------- host launcher ----------

extern "C" void kernel_launch(void* const* d_in, const int* in_sizes, int n_in,
                              void* d_out, int out_size, void* d_ws, size_t ws_size,
                              hipStream_t stream) {
  const float* edge_vector = (const float*)d_in[0];
  const int*   species     = (const int*)d_in[1];
  const int*   edge_index  = (const int*)d_in[2];
  const float* W_emb       = (const float*)d_in[3];
  const float* rad_w1      = (const float*)d_in[4];
  const float* rad_w2      = (const float*)d_in[5];
  const float* W_gate      = (const float*)d_in[6];
  const float* W_sc        = (const float*)d_in[7];
  const float* W_mix       = (const float*)d_in[8];
  int E = in_sizes[0] / 3;
  int N = in_sizes[1];
  float* out = (float*)d_out;
  (void)n_in; (void)out_size; (void)ws_size;

  int Mp_e = (E + 15) & ~15;
  int Mp_n = (N + 15) & ~15;

  char* p = (char*)d_ws;
  auto alloc = [&](size_t bytes) -> char* {
    char* r = p; p += (bytes + 255) & ~(size_t)255; return r;
  };
  float*  u      = (float*) alloc((size_t)E*3*sizeof(float));
  float*  ef     = (float*) alloc((size_t)E*NBASIS*sizeof(float));
  half_t* hidden = (half_t*)alloc((size_t)Mp_e*C_CH*sizeof(half_t));
  float*  rw     = (float*) alloc((size_t)Mp_e*RWN*sizeof(float));
  half_t* w2t    = (half_t*)alloc((size_t)2*256*64*sizeof(half_t));
  half_t* wgt    = (half_t*)alloc((size_t)2*64*64*sizeof(half_t));
  float*  h0a    = (float*) alloc((size_t)Mp_n*C_CH*sizeof(float));
  float*  h0b    = (float*) alloc((size_t)Mp_n*C_CH*sizeof(float));
  half_t* h0h    = (half_t*)alloc((size_t)Mp_n*C_CH*sizeof(half_t));
  float*  sgate  = (float*) alloc((size_t)Mp_n*C_CH*sizeof(float));
  float*  h1     = (float*) alloc((size_t)N*C_CH*3*sizeof(float));
  float*  h2     = (float*) alloc((size_t)N*C_CH*9*sizeof(float));
  float*  Aacc   = (float*) alloc((size_t)N*C_CH*ADIM*sizeof(float));

  auto cdiv = [](long long a, long long b) { return (int)((a + b - 1) / b); };

  // weight prep: f16-transposed B matrices for WMMA
  for (int l = 0; l < 2; l++) {
    k_transpose_f16<<<cdiv(64*256,256),256,0,stream>>>(rad_w2 + (size_t)l*64*256,
                                                       w2t + (size_t)l*256*64, 64, 256);
    k_transpose_f16<<<cdiv(64*64,256),256,0,stream>>>(W_gate + (size_t)l*64*64,
                                                      wgt + (size_t)l*64*64, 64, 64);
  }
  k_init_h0<<<cdiv((long long)N*C_CH,256),256,0,stream>>>(species, W_emb, h0a, N);
  k_edge_geom<<<cdiv(E,256),256,0,stream>>>(edge_vector, u, ef, E);

  float* h0cur = h0a; float* h0nxt = h0b;
  for (int layer = 0; layer < 2; layer++) {
    k_radial_hidden<<<cdiv((long long)E*C_CH,256),256,0,stream>>>(
        ef, rad_w1 + (size_t)layer*NBASIS*C_CH, hidden, E);
    {
      long long waves = (long long)(Mp_e/16) * (RWN/16);
      k_gemm_wmma<<<cdiv(waves,8),256,0,stream>>>(hidden, w2t + (size_t)layer*256*64,
                                                  rw, Mp_e, RWN, 64);
    }
    k_f32_to_f16<<<cdiv((long long)N*C_CH,256),256,0,stream>>>(h0cur, h0h, N*C_CH);
    {
      long long waves = (long long)(Mp_n/16) * (64/16);
      k_gemm_wmma<<<cdiv(waves,8),256,0,stream>>>(h0h, wgt + (size_t)layer*64*64,
                                                  sgate, Mp_n, 64, 64);
    }
    k_zero<<<cdiv((long long)N*C_CH*ADIM,256),256,0,stream>>>(Aacc, (long long)N*C_CH*ADIM);
    k_message<<<cdiv((long long)E*C_CH,256),256,0,stream>>>(
        rw, sgate, h1, h2, u, edge_index, Aacc, E, layer > 0 ? 1 : 0);
    k_node_update<<<cdiv(N,4),256,0,stream>>>(
        Aacc, h0cur, species, W_sc + (size_t)layer*ZMAX_*C_CH,
        W_mix + (size_t)layer*3*64*64, h0nxt, h1, h2, out, N,
        layer == 0 ? 1 : 0, layer*C_CH);
    float* t = h0cur; h0cur = h0nxt; h0nxt = t;
  }
}
